// DQN_24103356465496
// MI455X (gfx1250) — matrix operand order repaired and checked by reading, not compiled, not run on hardware
//
#include <hip/hip_runtime.h>
#include <hip/hip_bf16.h>
#include <cstdint>
#include <cstddef>

// Problem sizes (fixed by the reference)
enum : int { Bq = 8192, INq = 2048, Hq = 256, Aq = 8192, Kq = 512 };

typedef __attribute__((ext_vector_type(16))) __bf16   v16bf;
typedef __attribute__((ext_vector_type(8)))  __bf16   v8bf;
typedef __attribute__((ext_vector_type(8)))  float    v8f;
typedef __attribute__((ext_vector_type(4)))  uint32_t v4u;
typedef __attribute__((ext_vector_type(8)))  int      v8i;
typedef __attribute__((ext_vector_type(4)))  int      v4i;

// Block tiling
#define BM 128
#define BN 128
#define BK 32
#define LDS_STRIDE 40          // 32 bf16 + 8 pad (16B) -> 80B row pitch, 20-bank spacing
#define MAT_ELEMS  (128 * LDS_STRIDE)          // per matrix per buffer (5120 bf16 = 10240 B)
#define SMEM_BYTES (2 /*buf*/ * 4 /*Ahi,Alo,Bhi,Blo*/ * MAT_ELEMS * 2)   // 81920 B

// ---------- bf16 helpers (truncation split: hi + lo reconstructs ~fp32) ----------
__device__ __forceinline__ __bf16 bf_hi(float f) {
    union { float f; uint32_t u; } v; v.f = f;
    uint16_t h = (uint16_t)(v.u >> 16);
    __bf16 r; __builtin_memcpy(&r, &h, 2); return r;
}
__device__ __forceinline__ float bf2f(__bf16 b) {
    uint16_t h; __builtin_memcpy(&h, &b, 2);
    union { uint32_t u; float f; } v; v.u = ((uint32_t)h) << 16;
    return v.f;
}

// ---------- TDM descriptor build + issue (per CDNA5 ISA D# layout) ----------
// 2D tile of 2-byte elements: tile_d0 x tile_d1, row stride d0_stride (elements).
// pad_enable: insert 16B into LDS after every 64B stored -> 80B pitch.
__device__ __forceinline__ void tdm_load_2d_bf16(const void* lds_dst, const void* gsrc,
                                                 uint32_t tile_d0, uint32_t tile_d1,
                                                 uint64_t d0_stride) {
    uint64_t ga = (uint64_t)(uintptr_t)gsrc;
    v4u g0;
    g0[0] = 1u;                                   // count=1 (valid), user mode
    g0[1] = (uint32_t)(uintptr_t)lds_dst;         // LDS byte address (low 32 bits of flat addr)
    g0[2] = (uint32_t)(ga & 0xffffffffu);         // global_addr[31:0]
    g0[3] = (uint32_t)((ga >> 32) & 0x01ffffffu)  // global_addr[56:32]
          | (2u << 30);                           // type = 2 ("image")
    v8i g1;
    // data_size=1 (2B), pad_enable=1, pad_interval=3 (16 DW = 64B), pad_amount=3 (4 DW = 16B)
    g1[0] = (int)((1u << 16) | (1u << 20) | (3u << 22) | (3u << 25));
    g1[1] = (int)((tile_d0 & 0xffffu) << 16);                 // tensor_dim0[15:0]
    g1[2] = (int)(((tile_d0 >> 16) & 0xffffu) | ((tile_d1 & 0xffffu) << 16)); // t_d0[31:16], t_d1[15:0]
    g1[3] = (int)(((tile_d1 >> 16) & 0xffffu) | ((tile_d0 & 0xffffu) << 16)); // t_d1[31:16], tile_dim0
    g1[4] = (int)(tile_d1 & 0xffffu);                         // tile_dim1 (tile_dim2 = 0)
    g1[5] = (int)(uint32_t)(d0_stride & 0xffffffffu);         // tensor_dim0_stride[31:0]
    g1[6] = (int)(uint32_t)((d0_stride >> 32) & 0xffffu);     // stride[47:32], dim1_stride[15:0]=0
    g1[7] = 0;
    v4i z = {0, 0, 0, 0};
#if __clang_major__ >= 23
    v8i z8 = {0, 0, 0, 0, 0, 0, 0, 0};
    __builtin_amdgcn_tensor_load_to_lds(g0, g1, z, z, z8, 0);
#else
    __builtin_amdgcn_tensor_load_to_lds(g0, g1, z, z, 0);
#endif
}

// ---------- one-time conversion kernels (sources are read exactly once -> NT loads) ----------
__global__ void split_rm_kernel(const float* __restrict__ src,
                                __bf16* __restrict__ hi, __bf16* __restrict__ lo,
                                size_t n) {
    size_t i = blockIdx.x * (size_t)blockDim.x + threadIdx.x;
    if (i >= n) return;
    float f = __builtin_nontemporal_load(src + i);
    __bf16 h = bf_hi(f);
    hi[i] = h;
    lo[i] = bf_hi(f - bf2f(h));
}

// fp32 W[K,N] -> bf16 hi/lo packed as ((k/32)*N + n)*32 + (k&31):
// a 128-column B block for one 32-K slab is one contiguous 8 KB stream.
__global__ void split_packB_kernel(const float* __restrict__ w,
                                   __bf16* __restrict__ hi, __bf16* __restrict__ lo,
                                   int K, int N) {
    size_t i = blockIdx.x * (size_t)blockDim.x + threadIdx.x;
    if (i >= (size_t)K * N) return;
    int k = (int)(i / N);
    int n = (int)(i % N);
    float f = __builtin_nontemporal_load(w + i);
    __bf16 h = bf_hi(f);
    size_t d = ((size_t)(k >> 5) * N + n) * 32 + (k & 31);
    hi[d] = h;
    lo[d] = bf_hi(f - bf2f(h));
}

// ---------- legal-move bitmask ----------
__global__ void zero_mask_kernel(uint32_t* __restrict__ mask, size_t nwords) {
    size_t i = blockIdx.x * (size_t)blockDim.x + threadIdx.x;
    if (i < nwords) mask[i] = 0u;
}

__global__ void scatter_mask_kernel(const int* __restrict__ moves,
                                    uint32_t* __restrict__ mask) {
    size_t t = blockIdx.x * (size_t)blockDim.x + threadIdx.x;
    if (t >= (size_t)Bq * Kq) return;
    int r = (int)(t / Kq);
    int idx = moves[t];
    if (idx < 0 || idx >= Aq) return;
    atomicOr(&mask[(size_t)r * (Aq >> 5) + (idx >> 5)], 1u << (idx & 31));
}

// ---------- LDS-staged WMMA GEMM with TDM double buffering ----------
// Block: 256 threads (8 waves), 128x128 output tile, BK=32 per stage.
// Waves 0..3 each issue one TDM load per stage (A-hi, A-lo, B-hi, B-lo).
// Each wave computes a 64x32 sub-tile = 8 accumulators, bf16x3 per tile.
// MODE 0: relu, store bf16 hi/lo.  MODE 1: move-mask + (-inf), NT-store fp32.
template <int MODE>
__global__ __launch_bounds__(256)
void gemm_tdm_kernel(const __bf16* __restrict__ Ahi, const __bf16* __restrict__ Alo,
                     const __bf16* __restrict__ Bhi, const __bf16* __restrict__ Blo,
                     const float* __restrict__ bias, int K, int N,
                     __bf16* __restrict__ Ohi, __bf16* __restrict__ Olo,
                     float* __restrict__ Out, const uint32_t* __restrict__ mask) {
    extern __shared__ __bf16 smem[];
    // layout: [buf][mat: Ahi,Alo,Bhi,Blo][MAT_ELEMS]
    auto matp = [&](int buf, int mat) -> __bf16* { return smem + (size_t)(buf * 4 + mat) * MAT_ELEMS; };

    const int tid  = threadIdx.x;
    const int lane = tid & 31;
    const int wid  = tid >> 5;
    const int bm = blockIdx.y << 7;
    const int bn = blockIdx.x << 7;

    const int rc    = lane & 15;         // A row / B col within 16x16 tile
    const int kgrp  = (lane >> 4) << 3;  // A K sub-group: 0 or 8
    const int khalf = (lane >> 4) << 4;  // unused (B also uses kgrp): 0 or 16

    const int wm = (wid >> 2) << 6;      // wave M offset: 0 or 64
    const int wn = (wid & 3) << 5;       // wave N offset: 0,32,64,96

    auto stage = [&](int buf, int k0) {
        if (wid == 0)
            tdm_load_2d_bf16(matp(buf, 0), Ahi + (size_t)bm * K + k0, 32, 128, (uint64_t)K);
        else if (wid == 1)
            tdm_load_2d_bf16(matp(buf, 1), Alo + (size_t)bm * K + k0, 32, 128, (uint64_t)K);
        else if (wid == 2)
            tdm_load_2d_bf16(matp(buf, 2), Bhi + ((size_t)(k0 >> 5) * N + bn) * 32, 4096, 1, 4096);
        else if (wid == 3)
            tdm_load_2d_bf16(matp(buf, 3), Blo + ((size_t)(k0 >> 5) * N + bn) * 32, 4096, 1, 4096);
    };

    v8f acc[4][2] = {};
    const int steps = K >> 5;

    stage(0, 0);
    for (int s = 0; s < steps; ++s) {
        const int buf = s & 1;
        // scalar wait; waves that issued no TDM have TENSORcnt==0 already
        __builtin_amdgcn_s_wait_tensorcnt((short)0);
        __syncthreads();                              // buf is fully staged
        if (s + 1 < steps) stage(buf ^ 1, (s + 1) << 5);  // overlap next stage with compute

        const __bf16* sAh = matp(buf, 0);
        const __bf16* sAl = matp(buf, 1);
        const __bf16* sBh = matp(buf, 2);
        const __bf16* sBl = matp(buf, 3);

        v16bf bh[2], bl[2];
#pragma unroll
        for (int nt = 0; nt < 2; ++nt) {
            const int c = wn + (nt << 4) + rc;
            bh[nt] = __builtin_shufflevector(*(const v8bf*)(sBh + c * LDS_STRIDE + kgrp), *(const v8bf*)(sBh + c * LDS_STRIDE + 16 + kgrp), 0,1,2,3,4,5,6,7,8,9,10,11,12,13,14,15);
            bl[nt] = __builtin_shufflevector(*(const v8bf*)(sBl + c * LDS_STRIDE + kgrp), *(const v8bf*)(sBl + c * LDS_STRIDE + 16 + kgrp), 0,1,2,3,4,5,6,7,8,9,10,11,12,13,14,15);
        }
#pragma unroll
        for (int mt = 0; mt < 4; ++mt) {
            const int r = wm + (mt << 4) + rc;
            const __bf16* ph = sAh + r * LDS_STRIDE;
            const __bf16* pl = sAl + r * LDS_STRIDE;
            v8bf ah0 = *(const v8bf*)(ph + kgrp);
            v8bf ah1 = *(const v8bf*)(ph + 16 + kgrp);
            v8bf al0 = *(const v8bf*)(pl + kgrp);
            v8bf al1 = *(const v8bf*)(pl + 16 + kgrp);
            v16bf ahi = __builtin_shufflevector(ah0, ah1, 0,1,2,3,4,5,6,7,8,9,10,11,12,13,14,15);
            v16bf alo = __builtin_shufflevector(al0, al1, 0,1,2,3,4,5,6,7,8,9,10,11,12,13,14,15);
#pragma unroll
            for (int nt = 0; nt < 2; ++nt) {
                acc[mt][nt] = __builtin_amdgcn_wmma_f32_16x16x32_bf16(false, ahi, false, bh[nt], (short)0, acc[mt][nt], false, false);
                acc[mt][nt] = __builtin_amdgcn_wmma_f32_16x16x32_bf16(false, alo, false, bh[nt], (short)0, acc[mt][nt], false, false);
                acc[mt][nt] = __builtin_amdgcn_wmma_f32_16x16x32_bf16(false, ahi, false, bl[nt], (short)0, acc[mt][nt], false, false);
            }
        }
        __syncthreads();                              // done reading buf before it is restaged
    }

    // ---- epilogue ----
    float bnv[2];
#pragma unroll
    for (int nt = 0; nt < 2; ++nt) bnv[nt] = bias[bn + wn + (nt << 4) + rc];
    const int mrow0 = bm + wm + ((lane >> 4) << 3);

    if (MODE == 0) {
#pragma unroll
        for (int mt = 0; mt < 4; ++mt)
#pragma unroll
            for (int nt = 0; nt < 2; ++nt)
#pragma unroll
                for (int v = 0; v < 8; ++v) {
                    int m = mrow0 + (mt << 4) + v;
                    int n = bn + wn + (nt << 4) + rc;
                    float q = fmaxf(acc[mt][nt][v] + bnv[nt], 0.0f);
                    __bf16 h = bf_hi(q);
                    size_t idx = (size_t)m * N + n;
                    Ohi[idx] = h;                     // h1/h2 are re-read next layer: keep RT
                    Olo[idx] = bf_hi(q - bf2f(h));
                }
    } else {
        const int words = N >> 5;
#pragma unroll
        for (int mt = 0; mt < 4; ++mt)
#pragma unroll
            for (int nt = 0; nt < 2; ++nt)
#pragma unroll
                for (int v = 0; v < 8; ++v) {
                    int m = mrow0 + (mt << 4) + v;
                    int n = bn + wn + (nt << 4) + rc;
                    float q = acc[mt][nt][v] + bnv[nt];
                    uint32_t w = mask[(size_t)m * words + (n >> 5)];
                    bool legal = (w >> (n & 31)) & 1u;
                    // reference: filtered = q*mask; where(filtered==0) -> -inf
                    float r = (legal && q != 0.0f) ? q : -__builtin_inff();
                    // 256 MB, write-once, never re-read: NT store keeps W3/h2 resident in L2
                    __builtin_nontemporal_store(r, Out + (size_t)m * N + n);
                }
    }
}

extern "C" void kernel_launch(void* const* d_in, const int* in_sizes, int n_in,
                              void* d_out, int out_size, void* d_ws, size_t ws_size,
                              hipStream_t stream) {
    (void)in_sizes; (void)n_in; (void)out_size; (void)ws_size;
    const float* x  = (const float*)d_in[0];
    const int*   mv = (const int*)d_in[1];
    const float* W1 = (const float*)d_in[2];
    const float* b1 = (const float*)d_in[3];
    const float* W2 = (const float*)d_in[4];
    const float* b2 = (const float*)d_in[5];
    const float* W3 = (const float*)d_in[6];
    const float* b3 = (const float*)d_in[7];
    float* out = (float*)d_out;

    // ---- workspace carve-out (~99 MB total) ----
    char* ws = (char*)d_ws;
    size_t off = 0;
    auto alloc = [&](size_t bytes) -> void* {
        off = (off + 255) & ~(size_t)255;
        void* p = ws + off;
        off += bytes;
        return p;
    };
    uint32_t* mask = (uint32_t*)alloc((size_t)Bq * (Aq >> 5) * 4);
    __bf16* xhi  = (__bf16*)alloc((size_t)Bq * INq * 2);
    __bf16* xlo  = (__bf16*)alloc((size_t)Bq * INq * 2);
    __bf16* w1hi = (__bf16*)alloc((size_t)INq * Hq * 2);
    __bf16* w1lo = (__bf16*)alloc((size_t)INq * Hq * 2);
    __bf16* w2hi = (__bf16*)alloc((size_t)Hq * Hq * 2);
    __bf16* w2lo = (__bf16*)alloc((size_t)Hq * Hq * 2);
    __bf16* w3hi = (__bf16*)alloc((size_t)Hq * Aq * 2);
    __bf16* w3lo = (__bf16*)alloc((size_t)Hq * Aq * 2);
    __bf16* h1hi = (__bf16*)alloc((size_t)Bq * Hq * 2);
    __bf16* h1lo = (__bf16*)alloc((size_t)Bq * Hq * 2);
    __bf16* h2hi = (__bf16*)alloc((size_t)Bq * Hq * 2);
    __bf16* h2lo = (__bf16*)alloc((size_t)Bq * Hq * 2);

    const int T = 256;

    // 1) precision-split conversions + B-operand repack
    {
        size_t n = (size_t)Bq * INq;
        split_rm_kernel<<<dim3((unsigned)((n + T - 1) / T)), dim3(T), 0, stream>>>(x, xhi, xlo, n);
    }
    split_packB_kernel<<<dim3((unsigned)(((size_t)INq * Hq + T - 1) / T)), dim3(T), 0, stream>>>(W1, w1hi, w1lo, INq, Hq);
    split_packB_kernel<<<dim3((unsigned)(((size_t)Hq * Hq + T - 1) / T)), dim3(T), 0, stream>>>(W2, w2hi, w2lo, Hq, Hq);
    split_packB_kernel<<<dim3((unsigned)(((size_t)Hq * Aq + T - 1) / T)), dim3(T), 0, stream>>>(W3, w3hi, w3lo, Hq, Aq);

    // 2) legal-move bitmask
    {
        size_t nw = (size_t)Bq * (Aq >> 5);
        zero_mask_kernel<<<dim3((unsigned)((nw + T - 1) / T)), dim3(T), 0, stream>>>(mask, nw);
        size_t nt = (size_t)Bq * Kq;
        scatter_mask_kernel<<<dim3((unsigned)((nt + T - 1) / T)), dim3(T), 0, stream>>>(mv, mask);
    }

    // 3) h1 = relu(x @ W1 + b1)      [8192x2048 @ 2048x256]
    gemm_tdm_kernel<0><<<dim3(Hq / BN, Bq / BM), dim3(T), SMEM_BYTES, stream>>>(
        xhi, xlo, w1hi, w1lo, b1, INq, Hq, h1hi, h1lo, nullptr, nullptr);
    // 4) h2 = relu(h1 @ W2 + b2)     [8192x256 @ 256x256]
    gemm_tdm_kernel<0><<<dim3(Hq / BN, Bq / BM), dim3(T), SMEM_BYTES, stream>>>(
        h1hi, h1lo, w2hi, w2lo, b2, Hq, Hq, h2hi, h2lo, nullptr, nullptr);
    // 5) out = masked(h2 @ W3 + b3)  [8192x256 @ 256x8192]
    gemm_tdm_kernel<1><<<dim3(Aq / BN, Bq / BM), dim3(T), SMEM_BYTES, stream>>>(
        h2hi, h2lo, w3hi, w3lo, b3, Hq, Aq, nullptr, nullptr, out, mask);
}
